// GCN_Net_79242146611357
// MI455X (gfx1250) — matrix operand order repaired and checked by reading, not compiled, not run on hardware
//
#include <hip/hip_runtime.h>
#include <cstdint>
#include <cstring>

// Problem constants (match reference)
#define NNODES 100000
#define FEAT   128
#define HID    128
#define NCLS   64

typedef __attribute__((ext_vector_type(16))) __bf16 v16bf;
typedef __attribute__((ext_vector_type(8)))  float  v8f;

__device__ __forceinline__ unsigned short f2bf_bits(float f) {
    union { float f; unsigned int u; } x; x.f = f;
    unsigned int r = (x.u + 0x7FFFu + ((x.u >> 16) & 1u)) >> 16;  // RNE
    return (unsigned short)r;
}

// ---------------- degree / normalization ----------------

__global__ void deg_init_k(float* deg, int n) {
    int i = blockIdx.x * blockDim.x + threadIdx.x;
    if (i < n) deg[i] = 1.0f;  // self loop
}

__global__ void deg_accum_k(const int* __restrict__ dst, float* deg, int e) {
    int i = blockIdx.x * blockDim.x + threadIdx.x;
    if (i < e) atomicAdd(&deg[dst[i]], 1.0f);
}

__global__ void deg_to_dinv_k(float* d, int n) {
    int i = blockIdx.x * blockDim.x + threadIdx.x;
    if (i < n) d[i] = rsqrtf(d[i]);  // deg >= 1 always (self loop)
}

// ---------------- f32 -> bf16 conversion passes ----------------

// 4 floats per thread -> 4 packed bf16 (uint2)
__global__ void cvt_f32_bf16_k(const float* __restrict__ in,
                               unsigned short* __restrict__ out, int n4) {
    int i = blockIdx.x * blockDim.x + threadIdx.x;
    if (i >= n4) return;
    float4 v = reinterpret_cast<const float4*>(in)[i];
    uint2 p;
    p.x = (unsigned)f2bf_bits(v.x) | ((unsigned)f2bf_bits(v.y) << 16);
    p.y = (unsigned)f2bf_bits(v.z) | ((unsigned)f2bf_bits(v.w) << 16);
    reinterpret_cast<uint2*>(out)[i] = p;
}

// fused ReLU + f32 -> bf16 (produces layer-2 A operand)
__global__ void relu_bf16_k(const float* __restrict__ in,
                            unsigned short* __restrict__ out, int n4) {
    int i = blockIdx.x * blockDim.x + threadIdx.x;
    if (i >= n4) return;
    float4 v = reinterpret_cast<const float4*>(in)[i];
    v.x = fmaxf(v.x, 0.0f); v.y = fmaxf(v.y, 0.0f);
    v.z = fmaxf(v.z, 0.0f); v.w = fmaxf(v.w, 0.0f);
    uint2 p;
    p.x = (unsigned)f2bf_bits(v.x) | ((unsigned)f2bf_bits(v.y) << 16);
    p.y = (unsigned)f2bf_bits(v.z) | ((unsigned)f2bf_bits(v.w) << 16);
    reinterpret_cast<uint2*>(out)[i] = p;
}

// ---------------- WMMA GEMM: C[M,NOUT] = A[M,128](bf16) * W[128,NOUT](f32) ----------------
// One wave computes one 16x16 output tile with 4x v_wmma_f32_16x16x32_bf16.
// B tile (shared by all 8 waves of the block) is staged in LDS pre-packed in
// per-lane WMMA fragment order: Bsh[kb][lane][e], so each lane reads its whole
// 32-byte fragment with two ds_load_b128.

template <int NOUT>
__global__ void __launch_bounds__(256)
gemm_bf16_wmma_k(const unsigned short* __restrict__ A, const float* __restrict__ W,
                 float* __restrict__ C, int mTiles) {
    constexpr int K = 128;
    __shared__ alignas(16) unsigned short Bsh[4 * 32 * 16];  // [kb][lane][e], 4 KB

    const int nBase = blockIdx.y * 16;
    // stage + fragment-pack B: element (kb, lane, e) comes from W[k, nBase+n]
    // with hi = lane>>4, n = lane&15, k = kb*32 + (e>>3)*16 + hi*8 + (e&7)
    for (int i = threadIdx.x; i < 4 * 32 * 16; i += 256) {
        int kb   = i >> 9;
        int lane = (i >> 4) & 31;
        int e    = i & 15;
        int k    = kb * 32 + (e >> 3) * 16 + (lane >> 4) * 8 + (e & 7);
        int n    = lane & 15;
        Bsh[i] = f2bf_bits(W[k * NOUT + nBase + n]);
    }
    __syncthreads();

    const int wave  = threadIdx.x >> 5;
    const int lane  = threadIdx.x & 31;
    const int tileM = blockIdx.x * 8 + wave;
    if (tileM >= mTiles) return;

    const int hi   = lane >> 4;  // half-wave select
    const int mRow = tileM * 16 + (lane & 15);
    const uint4* Arow4 = reinterpret_cast<const uint4*>(A + (size_t)mRow * K); // 8 bf16 per uint4
    const uint4* Bfrag = reinterpret_cast<const uint4*>(&Bsh[lane * 16]);      // +kb*512 shorts

    v8f acc = {};
#pragma unroll
    for (int kb = 0; kb < 4; ++kb) {
        union { uint4 q[2]; v16bf v; } a, b;
        // A 16x32 bf16 fragment: K = kb*32 + hi*8 + {0..7} and +16
        a.q[0] = Arow4[kb * 4 + hi];
        a.q[1] = Arow4[kb * 4 + 2 + hi];
        // B fragment: prepacked, contiguous 32 bytes per lane
        b.q[0] = Bfrag[kb * 64 + 0];   // kb*512 shorts = kb*64 uint4
        b.q[1] = Bfrag[kb * 64 + 1];
        acc = __builtin_amdgcn_wmma_f32_16x16x32_bf16(
            false, a.v, false, b.v, (short)0, acc, false, false);
    }

    // C/D layout: vgpr j -> row (hi*8 + j), col (lane&15)
    float* Crow = C + (size_t)(tileM * 16 + hi * 8) * NOUT + nBase + (lane & 15);
#pragma unroll
    for (int j = 0; j < 8; ++j) Crow[(size_t)j * NOUT] = acc[j];
}

// ---------------- aggregation (scatter with symmetric norm) ----------------

// out[i,f] = bias[f] + Hin[i,f] * dinv[i]^2   (self-loop message + bias)
template <int F>
__global__ void agg_init_k(const float* __restrict__ Hin, const float* __restrict__ dinv,
                           const float* __restrict__ bias, float* __restrict__ out, int n) {
    constexpr int G = F / 4;
    int tid = blockIdx.x * blockDim.x + threadIdx.x;
    if (tid >= n * G) return;
    int i  = tid / G;
    int f4 = (tid - i * G) * 4;
    float dv = dinv[i];
    float s = dv * dv;
    float4 h = reinterpret_cast<const float4*>(Hin + (size_t)i * F + f4)[0];
    float4 bb = reinterpret_cast<const float4*>(bias + f4)[0];
    float4 o;
    o.x = bb.x + h.x * s; o.y = bb.y + h.y * s;
    o.z = bb.z + h.z * s; o.w = bb.w + h.w * s;
    reinterpret_cast<float4*>(out + (size_t)i * F + f4)[0] = o;
}

// out[dst,f] += Hin[src,f] * dinv[src] * dinv[dst]   (4 features per thread)
template <int F>
__global__ void agg_edges_k(const int* __restrict__ src, const int* __restrict__ dst,
                            const float* __restrict__ dinv, const float* __restrict__ Hin,
                            float* __restrict__ out, int e) {
    constexpr int G = F / 4;
    int tid = blockIdx.x * blockDim.x + threadIdx.x;
    if (tid >= e * G) return;
    int ed = tid / G;
    int f4 = (tid - ed * G) * 4;
    int s = src[ed];
    int d = dst[ed];
    float nv = dinv[s] * dinv[d];
    float4 h = reinterpret_cast<const float4*>(Hin + (size_t)s * F + f4)[0];
    float* o = out + (size_t)d * F + f4;
    atomicAdd(o + 0, h.x * nv);
    atomicAdd(o + 1, h.y * nv);
    atomicAdd(o + 2, h.z * nv);
    atomicAdd(o + 3, h.w * nv);
}

// ---------------- log-softmax over 64 classes, one wave per node ----------------

__global__ void __launch_bounds__(256) log_softmax_k(float* __restrict__ logits, int n) {
    int node = blockIdx.x * 8 + (threadIdx.x >> 5);
    if (node >= n) return;
    int lane = threadIdx.x & 31;
    float* row = logits + (size_t)node * NCLS;
    float a = row[lane];
    float b = row[lane + 32];
    float m = fmaxf(a, b);
#pragma unroll
    for (int off = 16; off >= 1; off >>= 1) m = fmaxf(m, __shfl_xor(m, off, 32));
    float s = __expf(a - m) + __expf(b - m);
#pragma unroll
    for (int off = 16; off >= 1; off >>= 1) s += __shfl_xor(s, off, 32);
    float ls = __logf(s) + m;
    row[lane]      = a - ls;
    row[lane + 32] = b - ls;
}

// ---------------- launcher ----------------

extern "C" void kernel_launch(void* const* d_in, const int* in_sizes, int n_in,
                              void* d_out, int out_size, void* d_ws, size_t ws_size,
                              hipStream_t stream) {
    const float* x  = (const float*)d_in[0];   // [N,128]
    const int*   ei = (const int*)d_in[1];     // [2,E]  (row0 = src, row1 = dst)
    const float* W1 = (const float*)d_in[2];   // [128,128]
    const float* b1 = (const float*)d_in[3];   // [128]
    const float* W2 = (const float*)d_in[4];   // [128,64]
    const float* b2 = (const float*)d_in[5];   // [64]
    float* out = (float*)d_out;                // [N,64]

    const int Nn = NNODES;
    const int E  = in_sizes[1] / 2;
    const int* src = ei;
    const int* dst = ei + E;

    // workspace layout
    float* dinv = (float*)d_ws;                                // N f32
    unsigned short* Abf = (unsigned short*)(dinv + Nn);        // N*128 bf16 (Xbf, then Hbf)
    float* H1  = (float*)(Abf + (size_t)Nn * FEAT);            // N*128 f32
    float* AGG = H1 + (size_t)Nn * FEAT;                       // N*128 f32

    const int B = 256;
    const int mTiles = Nn / 16;                                // 6250 exactly

    // 1) symmetric norm coefficients
    deg_init_k<<<(Nn + B - 1) / B, B, 0, stream>>>(dinv, Nn);
    deg_accum_k<<<(E + B - 1) / B, B, 0, stream>>>(dst, dinv, E);
    deg_to_dinv_k<<<(Nn + B - 1) / B, B, 0, stream>>>(dinv, Nn);

    // 2) Xbf = bf16(X);  H1 = Xbf @ W1  (WMMA bf16)
    {
        int n4 = Nn * FEAT / 4;
        cvt_f32_bf16_k<<<(n4 + B - 1) / B, B, 0, stream>>>(x, Abf, n4);
        dim3 grid((mTiles + 7) / 8, HID / 16);
        gemm_bf16_wmma_k<HID><<<grid, B, 0, stream>>>(Abf, W1, H1, mTiles);
    }

    // 3) AGG = b1 + selfloop + scatter(edges);  Abf <- bf16(relu(AGG))
    agg_init_k<FEAT><<<((Nn * FEAT / 4) + B - 1) / B, B, 0, stream>>>(H1, dinv, b1, AGG, Nn);
    agg_edges_k<FEAT><<<((E * (FEAT / 4)) + B - 1) / B, B, 0, stream>>>(src, dst, dinv, H1, AGG, E);
    relu_bf16_k<<<((Nn * FEAT / 4) + B - 1) / B, B, 0, stream>>>(AGG, Abf, Nn * FEAT / 4);

    // 4) H2 = Hbf @ W2 -> reuse H1 buffer (N*64 f32 fits)
    {
        dim3 grid((mTiles + 7) / 8, NCLS / 16);
        gemm_bf16_wmma_k<NCLS><<<grid, B, 0, stream>>>(Abf, W2, H1, mTiles);
    }

    // 5) logits accumulate directly in d_out
    agg_init_k<NCLS><<<((Nn * NCLS / 4) + B - 1) / B, B, 0, stream>>>(H1, dinv, b2, out, Nn);
    agg_edges_k<NCLS><<<((E * (NCLS / 4)) + B - 1) / B, B, 0, stream>>>(src, dst, dinv, H1, out, E);

    // 6) log_softmax in place
    log_softmax_k<<<(Nn + 7) / 8, B, 0, stream>>>(out, Nn);
}
